// Attention_32401233281139
// MI455X (gfx1250) — matrix operand order repaired and checked by reading, compiled, not run on hardware
//
#include <hip/hip_runtime.h>
#include <stdint.h>

// ---------------------------------------------------------------------------
// Multi-head attention, MI455X (gfx1250), wave32, WMMA f16->f32.
// B=8, N=1024, C=1024, H=16, D=64.
// ---------------------------------------------------------------------------

typedef _Float16 f16;
typedef __attribute__((ext_vector_type(16))) _Float16 v16h;
typedef __attribute__((ext_vector_type(8)))  float    v8f;

union Frag16 { uint32_t u[8]; v16h h; };
union F16x2  { uint32_t u; _Float16 h[2]; };

__device__ __forceinline__ v8f wmma16(const Frag16& a, const Frag16& b, v8f c) {
    // D = A(16x32 f16) * B(32x16 f16) + C(16x16 f32)
    return __builtin_amdgcn_wmma_f32_16x16x32_f16(
        false, a.h, false, b.h, (short)0, c, false, false);
}

// k-offset for 16-bit A-matrix 16x32 fragment (per CDNA5 ISA 7.12.2):
// v0..3: K = half*8 + 2v ; v4..7: K = 16 + half*8 + 2(v-4)
__device__ __forceinline__ int a_kmap(int v, int half) {
    return (v < 4) ? (half * 8 + 2 * v) : (16 + half * 8 + 2 * (v - 4));
}
// B-matrix 32x16: same K order as A: v0..3: K = half*8 + 2v ; v4..7: K = 16 + half*8 + 2(v-4)
__device__ __forceinline__ int b_kmap(int v, int half) {
    return (v < 4) ? (half * 8 + 2 * v) : (16 + half * 8 + 2 * (v - 4));
}

// ---------------------------------------------------------------------------
// fp32 -> f16 conversion
// ---------------------------------------------------------------------------
__global__ void cvt_f32_to_f16(const float* __restrict__ src,
                               f16* __restrict__ dst, int n) {
    int i = blockIdx.x * blockDim.x + threadIdx.x;
    if (i < n) dst[i] = (f16)src[i];
}

// ---------------------------------------------------------------------------
// Block GEMM: Y[M=8192, N=1024] = X(f16) @ W(f16) + bias(f32)
// Block tile 128(M) x 256(N); 256 threads = 8 waves in a 2(M) x 4(N) grid;
// each wave computes a square 64x64 tile = 4x4 WMMA accumulators, so
// fragment LDS traffic is 8*(4+4)=64 dwords per 16 WMMA (4 dwords/WMMA).
// TO_HEADS=true  : write f16 to dst in [B,H,N,D] layout (QKV projections)
// TO_HEADS=false : write f32 to dst row-major [M,N]     (output projection)
// ---------------------------------------------------------------------------
template <bool TO_HEADS>
__global__ __launch_bounds__(256) void gemm128(
    const f16* __restrict__ X, const f16* __restrict__ W,
    const float* __restrict__ bias, f16* __restrict__ dstH,
    float* __restrict__ dstF) {
    constexpr int KDIM = 1024;
    constexpr int BM = 128, BN = 256, BK = 32;
    constexpr int LDA = BK + 8;  // f16 units, pad to dodge bank conflicts
    constexpr int LDB = BK + 8;

    __shared__ f16 sA[BM * LDA];   // A tile, row-major [m][k]   (10.2 KB)
    __shared__ f16 sB[BN * LDB];   // B tile, transposed [n][k]  (20.5 KB)

    const int tid  = threadIdx.x;
    const int lane = tid & 31;
    const int wid  = tid >> 5;
    const int half = lane >> 4;
    const int r    = lane & 15;
    const int wm   = wid >> 2;      // 0..1 -> 64-row slabs
    const int wn   = wid & 3;       // 0..3 -> 64-col slabs
    const int mBase = blockIdx.y * BM;
    const int nBase = blockIdx.x * BN;

    v8f acc[4][4] = {};

    for (int kt = 0; kt < KDIM / BK; ++kt) {
        const int kBase = kt * BK;
        if (kt + 1 < KDIM / BK) {   // prefetch next tiles toward L2
            __builtin_prefetch(&X[(size_t)(mBase + (tid >> 1)) * KDIM + kBase + BK], 0, 1);
            __builtin_prefetch(&W[(size_t)(kBase + BK + (tid >> 3)) * 1024 + nBase], 0, 1);
        }
        // stage A: 128x32 f16 = 2048 dwords / 256 threads = 8 each
#pragma unroll
        for (int i = 0; i < 8; ++i) {
            int idx = tid + i * 256;
            int row = idx >> 4, kp = (idx & 15) << 1;
            *(uint32_t*)&sA[row * LDA + kp] =
                *(const uint32_t*)&X[(size_t)(mBase + row) * KDIM + kBase + kp];
        }
        // stage B transposed: sB[n][k] = W[kBase+k][nBase+n]; 4096 dwords
#pragma unroll
        for (int i = 0; i < 16; ++i) {
            int idx = tid + i * 256;
            int k = idx >> 7, np = (idx & 127) << 1;
            F16x2 w2;
            w2.u = *(const uint32_t*)&W[(size_t)(kBase + k) * 1024 + nBase + np];
            sB[(np + 0) * LDB + k] = w2.h[0];
            sB[(np + 1) * LDB + k] = w2.h[1];
        }
        __syncthreads();

        Frag16 a[4];
#pragma unroll
        for (int mt = 0; mt < 4; ++mt) {
            int row = wm * 64 + mt * 16 + r;
#pragma unroll
            for (int v = 0; v < 8; ++v)
                a[mt].u[v] = *(const uint32_t*)&sA[row * LDA + a_kmap(v, half)];
        }
#pragma unroll
        for (int nt = 0; nt < 4; ++nt) {
            int col = wn * 64 + nt * 16 + r;
            Frag16 b;
#pragma unroll
            for (int v = 0; v < 8; ++v)
                b.u[v] = *(const uint32_t*)&sB[col * LDB + b_kmap(v, half)];
#pragma unroll
            for (int mt = 0; mt < 4; ++mt)
                acc[mt][nt] = wmma16(a[mt], b, acc[mt][nt]);
        }
        __syncthreads();
    }

    // epilogue: C layout -> lanes 0-15 row=g, lanes 16-31 row=g+8; col = lane&15
#pragma unroll
    for (int nt = 0; nt < 4; ++nt) {
        int col = nBase + wn * 64 + nt * 16 + r;
        float bv = bias[col];
#pragma unroll
        for (int mt = 0; mt < 4; ++mt) {
#pragma unroll
            for (int g = 0; g < 8; ++g) {
                int row = mBase + wm * 64 + mt * 16 + g + 8 * half;
                float val = acc[mt][nt][g] + bv;
                if (TO_HEADS) {
                    int bi = row >> 10, t = row & 1023;
                    int h = col >> 6, d = col & 63;
                    dstH[(((size_t)bi * 16 + h) * 1024 + t) * 64 + d] = (f16)val;
                } else {
                    dstF[(size_t)row * 1024 + col] = val;
                }
            }
        }
    }
}

// ---------------------------------------------------------------------------
// Flash attention per (b,h). Block = 128 query rows, 8 waves x 16 rows.
// Key loop in tiles of 64. Online softmax via __shfl_xor row reductions.
// ---------------------------------------------------------------------------
__global__ __launch_bounds__(256) void attn_kernel(
    const f16* __restrict__ Q, const f16* __restrict__ K,
    const f16* __restrict__ V, f16* __restrict__ AO) {
    constexpr int LDK = 68;  // 64 + 4 pad

    __shared__ f16 sK[64 * LDK];        // [key][d]
    __shared__ f16 sVt[64 * LDK];       // [d][key] (transposed)
    __shared__ f16 sP[8 * 16 * LDK];    // per-wave P tiles [row][key]

    const int tid  = threadIdx.x;
    const int lane = tid & 31;
    const int wid  = tid >> 5;
    const int half = lane >> 4;
    const int r    = lane & 15;
    const int bh   = blockIdx.y;                 // b*16 + h
    const size_t hdBase = (size_t)bh * 1024 * 64;
    const f16* Qp = Q + hdBase;
    const f16* Kp = K + hdBase;
    const f16* Vp = V + hdBase;
    const int q0 = blockIdx.x * 128 + wid * 16;  // this wave's query base

    // Q fragments for d = 0..63 (two K-steps of 32)
    Frag16 aq[2];
#pragma unroll
    for (int kbi = 0; kbi < 2; ++kbi)
#pragma unroll
        for (int v = 0; v < 8; ++v) {
            int d = kbi * 32 + a_kmap(v, half);
            aq[kbi].u[v] = *(const uint32_t*)&Qp[(size_t)(q0 + r) * 64 + d];
        }

    v8f o[4] = {};
    float m[8], l[8];
#pragma unroll
    for (int g = 0; g < 8; ++g) { m[g] = -1e30f; l[g] = 0.0f; }

    for (int kt = 0; kt < 16; ++kt) {
        const int kb0 = kt * 64;
        // stage K (row-major) and V (transposed)
#pragma unroll
        for (int i = 0; i < 8; ++i) {
            int idx = tid + i * 256;
            int row = idx >> 5, dp = (idx & 31) << 1;
            *(uint32_t*)&sK[row * LDK + dp] =
                *(const uint32_t*)&Kp[(size_t)(kb0 + row) * 64 + dp];
            F16x2 v2;
            v2.u = *(const uint32_t*)&Vp[(size_t)(kb0 + row) * 64 + dp];
            sVt[(dp + 0) * LDK + row] = v2.h[0];
            sVt[(dp + 1) * LDK + row] = v2.h[1];
        }
        __syncthreads();

        // S = (Q K^T) * 1/sqrt(64); B-matrix of K^T is exactly row-major sK
        v8f st[4];
#pragma unroll
        for (int nt = 0; nt < 4; ++nt) {
            v8f s = {};
#pragma unroll
            for (int kbi = 0; kbi < 2; ++kbi) {
                Frag16 b;
#pragma unroll
                for (int v = 0; v < 8; ++v)
                    b.u[v] = *(const uint32_t*)
                        &sK[(nt * 16 + r) * LDK + kbi * 32 + b_kmap(v, half)];
                s = wmma16(aq[kbi], b, s);
            }
#pragma unroll
            for (int e = 0; e < 8; ++e) s[e] *= 0.125f;
            st[nt] = s;
        }

        // row max over 64 keys (elementwise over tiles, then across 16 lanes)
        float mloc[8];
#pragma unroll
        for (int g = 0; g < 8; ++g) {
            float v = st[0][g];
#pragma unroll
            for (int nt = 1; nt < 4; ++nt) v = fmaxf(v, st[nt][g]);
            v = fmaxf(v, __shfl_xor(v, 1, 32));
            v = fmaxf(v, __shfl_xor(v, 2, 32));
            v = fmaxf(v, __shfl_xor(v, 4, 32));
            v = fmaxf(v, __shfl_xor(v, 8, 32));
            mloc[g] = v;
        }
        float alpha[8];
#pragma unroll
        for (int g = 0; g < 8; ++g) {
            float mn = fmaxf(m[g], mloc[g]);
            alpha[g] = __expf(m[g] - mn);
            m[g] = mn;
        }
        // P = exp(S - m); row sums
        v8f pt[4];
        float rs[8];
#pragma unroll
        for (int g = 0; g < 8; ++g) rs[g] = 0.0f;
#pragma unroll
        for (int nt = 0; nt < 4; ++nt)
#pragma unroll
            for (int g = 0; g < 8; ++g) {
                float p = __expf(st[nt][g] - m[g]);
                pt[nt][g] = p;
                rs[g] += p;
            }
#pragma unroll
        for (int g = 0; g < 8; ++g) {
            float v = rs[g];
            v += __shfl_xor(v, 1, 32);
            v += __shfl_xor(v, 2, 32);
            v += __shfl_xor(v, 4, 32);
            v += __shfl_xor(v, 8, 32);
            l[g] = l[g] * alpha[g] + v;
        }
        // rescale O
#pragma unroll
        for (int dt = 0; dt < 4; ++dt)
#pragma unroll
            for (int g = 0; g < 8; ++g) o[dt][g] *= alpha[g];

        // write P (C layout) to LDS, reload as A layout
        const int wbase = wid * 16 * LDK;
#pragma unroll
        for (int nt = 0; nt < 4; ++nt)
#pragma unroll
            for (int g = 0; g < 8; ++g)
                sP[wbase + (g + 8 * half) * LDK + nt * 16 + r] = (f16)pt[nt][g];
        __syncthreads();

        Frag16 ap[2];
#pragma unroll
        for (int kbi = 0; kbi < 2; ++kbi)
#pragma unroll
            for (int v = 0; v < 8; ++v)
                ap[kbi].u[v] = *(const uint32_t*)
                    &sP[wbase + r * LDK + kbi * 32 + a_kmap(v, half)];

        // O += P @ V   (B-matrix of V needs [d][key] -> sVt contiguous)
#pragma unroll
        for (int dt = 0; dt < 4; ++dt)
#pragma unroll
            for (int kbi = 0; kbi < 2; ++kbi) {
                Frag16 b;
#pragma unroll
                for (int v = 0; v < 8; ++v)
                    b.u[v] = *(const uint32_t*)
                        &sVt[(dt * 16 + r) * LDK + kbi * 32 + b_kmap(v, half)];
                o[dt] = wmma16(ap[kbi], b, o[dt]);
            }
        __syncthreads();  // before next tile overwrites sK/sVt
    }

    // epilogue: normalize and write attn-out in [B, N, C] (C = h*64 + d), f16
    const int bi = bh >> 4, h = bh & 15;
#pragma unroll
    for (int dt = 0; dt < 4; ++dt)
#pragma unroll
        for (int g = 0; g < 8; ++g) {
            int row = q0 + g + 8 * half;
            int col = dt * 16 + r;
            float val = o[dt][g] / l[g];
            AO[((size_t)bi * 1024 + row) * 1024 + h * 64 + col] = (f16)val;
        }
}

// ---------------------------------------------------------------------------
// Host launcher
// ---------------------------------------------------------------------------
extern "C" void kernel_launch(void* const* d_in, const int* in_sizes, int n_in,
                              void* d_out, int out_size, void* d_ws,
                              size_t ws_size, hipStream_t stream) {
    (void)in_sizes; (void)n_in; (void)out_size; (void)ws_size;
    const float* x  = (const float*)d_in[0];
    const float* Wq = (const float*)d_in[1];
    const float* bq = (const float*)d_in[2];
    const float* Wk = (const float*)d_in[3];
    const float* bk = (const float*)d_in[4];
    const float* Wv = (const float*)d_in[5];
    const float* bv = (const float*)d_in[6];
    const float* Wo = (const float*)d_in[7];
    const float* bo = (const float*)d_in[8];

    char* ws = (char*)d_ws;
    const size_t MB = 1u << 20;
    f16* xh  = (f16*)(ws + 0 * MB);    // 16 MB  (8192x1024)
    f16* wqh = (f16*)(ws + 16 * MB);   //  2 MB
    f16* wkh = (f16*)(ws + 18 * MB);
    f16* wvh = (f16*)(ws + 20 * MB);
    f16* woh = (f16*)(ws + 22 * MB);
    f16* qh  = (f16*)(ws + 24 * MB);   // 16 MB [B,H,N,D]
    f16* kh  = (f16*)(ws + 40 * MB);
    f16* vh  = (f16*)(ws + 56 * MB);
    f16* aoh = (f16*)(ws + 72 * MB);   // 16 MB [B,N,C]   (total 88 MB)

    const int NX = 8 * 1024 * 1024, NW = 1024 * 1024;
    cvt_f32_to_f16<<<(NX + 255) / 256, 256, 0, stream>>>(x, xh, NX);
    cvt_f32_to_f16<<<(NW + 255) / 256, 256, 0, stream>>>(Wq, wqh, NW);
    cvt_f32_to_f16<<<(NW + 255) / 256, 256, 0, stream>>>(Wk, wkh, NW);
    cvt_f32_to_f16<<<(NW + 255) / 256, 256, 0, stream>>>(Wv, wvh, NW);
    cvt_f32_to_f16<<<(NW + 255) / 256, 256, 0, stream>>>(Wo, woh, NW);

    dim3 gg(1024 / 256, 8192 / 128);   // (N tiles, M tiles)
    gemm128<true><<<gg, 256, 0, stream>>>(xh, wqh, bq, qh, nullptr);
    gemm128<true><<<gg, 256, 0, stream>>>(xh, wkh, bk, kh, nullptr);
    gemm128<true><<<gg, 256, 0, stream>>>(xh, wvh, bv, vh, nullptr);

    attn_kernel<<<dim3(1024 / 128, 8 * 16), 256, 0, stream>>>(qh, kh, vh, aoh);

    gemm128<false><<<gg, 256, 0, stream>>>(aoh, woh, bo, nullptr, (float*)d_out);
}
